// DocEEGNNModelHN_33423435498394
// MI455X (gfx1250) — matrix operand order repaired and checked by reading, not compiled, not run on hardware
//
#include <hip/hip_runtime.h>
#include <hip/hip_bf16.h>

// Problem constants (match reference)
#define N_NODES 100000
#define DIM     128
#define NEDGE   600000
#define NREL    5
#define ROWT    (N_NODES/16)   // 6250 row tiles, exact

typedef __attribute__((ext_vector_type(16))) __bf16       v16bf;
typedef __attribute__((ext_vector_type(8)))  float        v8f;
typedef __attribute__((ext_vector_type(8)))  unsigned int v8u;

__device__ __forceinline__ unsigned f2bf(float f) {
    union { float f; unsigned u; } c; c.f = f;
    unsigned u = c.u;
    return (u + 0x7fffu + ((u >> 16) & 1u)) >> 16;   // round-to-nearest-even
}

// ---------------------------------------------------------------------------
// Pack weight matrix [nmat][128][ncols] fp32 -> bf16 B-fragment stream.
// B packing used here (contiguous k, same as A): lane = N (lo/hi 16), VGPR j: K=2j,2j+1 (+16 hi lanes)
// grid = (4*ncols/16, nmat), block = 32
// ---------------------------------------------------------------------------
__global__ void pack_w_kernel(const float* __restrict__ src,
                              unsigned* __restrict__ dst, int ncols) {
    int NT   = ncols >> 4;
    int t    = blockIdx.x, mat = blockIdx.y, lane = (int)threadIdx.x;
    int ko   = t / NT, nt = t - ko * NT;
    int n    = nt * 16 + (lane & 15);
    int kb   = ko * 32 + ((lane >> 4) << 4);
    const float* s = src + (size_t)mat * 128 * ncols;
    unsigned* d = dst + (((size_t)(mat * gridDim.x + t)) * 32 + lane) * 8;
#pragma unroll
    for (int j = 0; j < 8; j++) {
        int k = kb + 2 * j;
        d[j] = f2bf(s[(size_t)k * ncols + n]) |
               (f2bf(s[(size_t)(k + 1) * ncols + n]) << 16);
    }
}

// ---------------------------------------------------------------------------
// Pack activations [N][128] fp32 -> bf16 A-fragment stream (optional exact GELU).
// A packing used here (contiguous k, same as B): lane<16: M=lane, K in {0..15}; lane>=16: +16
// grid = (ROWT, 4), block = 32
// ---------------------------------------------------------------------------
__global__ void pack_a_kernel(const float* __restrict__ src,
                              unsigned* __restrict__ dst, int do_gelu) {
    int rb = blockIdx.x, ko = blockIdx.y, lane = (int)threadIdx.x;
    int m  = rb * 16 + (lane & 15);
    int kb = ko * 32 + ((lane >> 4) << 4);
    const float* s = src + (size_t)m * DIM;
    unsigned* d = dst + (((size_t)(rb * 4 + ko)) * 32 + lane) * 8;
#pragma unroll
    for (int j = 0; j < 8; j++) {
        int k = kb + 2 * j;
        float a = s[k], b = s[k + 1];
        if (do_gelu) {
            a = 0.5f * a * (1.0f + erff(a * 0.7071067811865475f));
            b = 0.5f * b * (1.0f + erff(b * 0.7071067811865475f));
        }
        d[j] = f2bf(a) | (f2bf(b) << 16);
    }
}

// ---------------------------------------------------------------------------
// GEMM: C[N x ntfull*16] = A[N x 128] * W[128 x ntfull*16], bf16 in / f32 out.
// One wave per (16-row block, 128-col group). All 4 A fragments preloaded;
// B fragments double-buffered across the 8 column tiles so loads for tile
// nt+1 are in flight while tile nt's 4 WMMAs execute. Optional fused bias.
// grid = (ROWT, ntfull/8), block = 32 (one wave).
// ---------------------------------------------------------------------------
__global__ __launch_bounds__(32)
void gemm8_kernel(const unsigned* __restrict__ Afrag,
                  const unsigned* __restrict__ Wfrag,
                  const float* __restrict__ bias,
                  float* __restrict__ C, int ntfull) {
    int rb = blockIdx.x, cg = blockIdx.y, lane = (int)threadIdx.x;
    const v8u* A = (const v8u*)Afrag;
    const v8u* B = (const v8u*)Wfrag;

    // preload all four 16x32 A fragments for this row block (32 VGPRs)
    v16bf a[4];
#pragma unroll
    for (int ko = 0; ko < 4; ko++)
        a[ko] = __builtin_bit_cast(v16bf, A[(size_t)(rb * 4 + ko) * 32 + lane]);

    v8f acc[8];
#pragma unroll
    for (int i = 0; i < 8; i++) acc[i] = (v8f)0.0f;

    // double-buffered B fragments: bb[buf][ko]
    v8u bb[2][4];
#pragma unroll
    for (int ko = 0; ko < 4; ko++)
        bb[0][ko] = B[(size_t)(ko * ntfull + cg * 8) * 32 + lane];

#pragma unroll
    for (int nt = 0; nt < 8; nt++) {
        int cur = nt & 1;
        if (nt < 7) {
#pragma unroll
            for (int ko = 0; ko < 4; ko++)
                bb[cur ^ 1][ko] =
                    B[(size_t)(ko * ntfull + cg * 8 + nt + 1) * 32 + lane];
        }
#pragma unroll
        for (int ko = 0; ko < 4; ko++)
            acc[nt] = __builtin_amdgcn_wmma_f32_16x16x32_bf16(
                false, a[ko], false, __builtin_bit_cast(v16bf, bb[cur][ko]),
                (short)0, acc[nt], false, false);
    }

    // C/D layout: lane<16 -> rows rb*16+0..7, col=lane; lane>=16 -> rows +8
    int cl = lane & 15;
    int rbase = rb * 16 + ((lane >> 4) << 3);
    int ldc = ntfull * 16;
    int colbase = cg * 128;
#pragma unroll
    for (int nt = 0; nt < 8; nt++) {
        int col = colbase + nt * 16 + cl;
        float bv = bias ? bias[col] : 0.0f;
#pragma unroll
        for (int v = 0; v < 8; v++)
            C[(size_t)(rbase + v) * ldc + col] = acc[nt][v] + bv;
    }
}

// out = relu(gamma_s * xs + beta_s) with bg = [beta | gamma] (N x 256)
__global__ void skip_combine_kernel(const float* __restrict__ bg,
                                    const float* __restrict__ xs,
                                    float* __restrict__ out) {
    const int total = N_NODES * DIM;
    for (int i = blockIdx.x * blockDim.x + threadIdx.x; i < total;
         i += gridDim.x * blockDim.x) {
        int n = i >> 7, c = i & 127;
        float v = bg[(size_t)n * 256 + 128 + c] * xs[i] + bg[(size_t)n * 256 + c];
        out[i] = v > 0.0f ? v : 0.0f;
    }
}

__global__ void zero_kernel(float* __restrict__ p, int count) {
    for (int i = blockIdx.x * blockDim.x + threadIdx.x; i < count;
         i += gridDim.x * blockDim.x) p[i] = 0.0f;
}

// One wave per edge of matching type: msg = relu(gamma[dst]*xl[src]+beta[dst]),
// atomically accumulated into summed[dst], cnt[dst]. block = (32,8) -> 8 edges
__global__ void edge_kernel(const int* __restrict__ ei, const int* __restrict__ et,
                            const float* __restrict__ bg, const float* __restrict__ xl,
                            float* __restrict__ summed, float* __restrict__ cnt,
                            int rel) {
    int e = blockIdx.x * 8 + threadIdx.y;
    if (e >= NEDGE) return;
    if (et[e] != rel) return;            // whole wave exits together
    int src = ei[e], dst = ei[NEDGE + e];
    int lane = (int)threadIdx.x;
    const float4* xr = (const float4*)(xl + (size_t)src * DIM);
    const float4* br = (const float4*)(bg + (size_t)dst * 256);
    float4 v = xr[lane];
    float4 be = br[lane];
    float4 ga = br[32 + lane];
    float m0 = fmaxf(ga.x * v.x + be.x, 0.0f);
    float m1 = fmaxf(ga.y * v.y + be.y, 0.0f);
    float m2 = fmaxf(ga.z * v.z + be.z, 0.0f);
    float m3 = fmaxf(ga.w * v.w + be.w, 0.0f);
    float* srow = summed + (size_t)dst * DIM + lane * 4;
    atomicAdd(srow + 0, m0);
    atomicAdd(srow + 1, m1);
    atomicAdd(srow + 2, m2);
    atomicAdd(srow + 3, m3);
    if (lane == 0) atomicAdd(cnt + dst, 1.0f);
}

__global__ void accum_kernel(const float* __restrict__ summed,
                             const float* __restrict__ cnt,
                             float* __restrict__ out) {
    const int total = N_NODES * DIM;
    for (int i = blockIdx.x * blockDim.x + threadIdx.x; i < total;
         i += gridDim.x * blockDim.x)
        out[i] += summed[i] / fmaxf(cnt[i >> 7], 1.0f);
}

extern "C" void kernel_launch(void* const* d_in, const int* in_sizes, int n_in,
                              void* d_out, int out_size, void* d_ws, size_t ws_size,
                              hipStream_t stream) {
    const float* x           = (const float*)d_in[0];
    const int*   edge_index  = (const int*)d_in[1];
    const int*   edge_type   = (const int*)d_in[2];
    const float* lin_skip_w  = (const float*)d_in[3];
    const float* film_skip_w = (const float*)d_in[4];
    const float* lins_w      = (const float*)d_in[5];
    const float* films_w     = (const float*)d_in[6];
    const float* films_b     = (const float*)d_in[7];
    const float* linear1_w   = (const float*)d_in[8];
    const float* linear1_b   = (const float*)d_in[9];
    float* outp = (float*)d_out;

    // workspace layout (all 256B aligned)
    char* ws = (char*)d_ws;
    size_t off = 0;
    auto take = [&](size_t bytes) {
        char* p = ws + off;
        off = (off + bytes + 255) & ~(size_t)255;
        return p;
    };
    unsigned* xfrag = (unsigned*)take((size_t)ROWT * 4 * 32 * 32);       // 25.6 MB
    unsigned* hfrag = (unsigned*)take((size_t)ROWT * 4 * 32 * 32);       // 25.6 MB
    unsigned* wfrag = (unsigned*)take((size_t)608 * 1024);               // 608 tiles
    float* bg     = (float*)take((size_t)N_NODES * 256 * 4);             // 102.4 MB
    float* xl     = (float*)take((size_t)N_NODES * DIM * 4);             // 51.2 MB
    float* outacc = (float*)take((size_t)N_NODES * DIM * 4);             // 51.2 MB
    float* summed = (float*)take((size_t)(N_NODES * DIM + N_NODES) * 4); // +cnt
    float* cnt    = summed + (size_t)N_NODES * DIM;

    // weight fragment sub-blocks (tile record = 256 uints)
    unsigned* wf_filmskip = wfrag;                 // 64 tiles (K=128,N=256)
    unsigned* wf_linskip  = wfrag + 64 * 256;      // 32 tiles
    unsigned* wf_lins     = wfrag + 96 * 256;      // 5 x 32 tiles
    unsigned* wf_films    = wfrag + 256 * 256;     // 5 x 64 tiles
    unsigned* wf_linear1  = wfrag + 576 * 256;     // 32 tiles

    // 1) pack weights + activations to bf16 WMMA fragments
    pack_w_kernel<<<dim3(64, 1), 32, 0, stream>>>(film_skip_w, wf_filmskip, 256);
    pack_w_kernel<<<dim3(32, 1), 32, 0, stream>>>(lin_skip_w,  wf_linskip,  128);
    pack_w_kernel<<<dim3(32, NREL), 32, 0, stream>>>(lins_w,   wf_lins,     128);
    pack_w_kernel<<<dim3(64, NREL), 32, 0, stream>>>(films_w,  wf_films,    256);
    pack_w_kernel<<<dim3(32, 1), 32, 0, stream>>>(linear1_w,  wf_linear1,  128);
    pack_a_kernel<<<dim3(ROWT, 4), 32, 0, stream>>>(x, xfrag, 0);

    // 2) skip path: out = relu(gamma_s * (x@lin_skip_w) + beta_s)
    gemm8_kernel<<<dim3(ROWT, 2), 32, 0, stream>>>(xfrag, wf_filmskip, nullptr, bg, 16);
    gemm8_kernel<<<dim3(ROWT, 1), 32, 0, stream>>>(xfrag, wf_linskip, nullptr, xl, 8);
    skip_combine_kernel<<<2048, 256, 0, stream>>>(bg, xl, outacc);

    // 3) per-relation message passing (each edge processed once, by its type)
    for (int r = 0; r < NREL; r++) {
        gemm8_kernel<<<dim3(ROWT, 2), 32, 0, stream>>>(
            xfrag, wf_films + (size_t)r * 64 * 256, films_b + r * 256, bg, 16);
        gemm8_kernel<<<dim3(ROWT, 1), 32, 0, stream>>>(
            xfrag, wf_lins + (size_t)r * 32 * 256, nullptr, xl, 8);
        zero_kernel<<<2048, 256, 0, stream>>>(summed, N_NODES * DIM + N_NODES);
        edge_kernel<<<NEDGE / 8, dim3(32, 8), 0, stream>>>(
            edge_index, edge_type, bg, xl, summed, cnt, r);
        accum_kernel<<<2048, 256, 0, stream>>>(summed, cnt, outacc);
    }

    // 4) final: gelu(out) @ linear1_w + linear1_b  (gelu fused into A-pack)
    pack_a_kernel<<<dim3(ROWT, 4), 32, 0, stream>>>(outacc, hfrag, 1);
    gemm8_kernel<<<dim3(ROWT, 1), 32, 0, stream>>>(hfrag, wf_linear1, linear1_b, outp, 8);
}
